// DecoderAttentionHead_32366873543152
// MI455X (gfx1250) — matrix operand order repaired and checked by reading, not compiled, not run on hardware
//
#include <hip/hip_runtime.h>
#include <hip/hip_bf16.h>
#include <math.h>

// ---------------------------------------------------------------------------
// CDNA5 (gfx1250) fused decoder attention head, bf16 WMMA everywhere.
//   d = 128, S = 2048, B = 8, E = 1024, O = 1024
// Pipeline:
//   1) q/k/v = x @ W + b          (WMMA bf16 GEMM, bf16 outputs in ws)
//   2) flash attention (causal)   (WMMA bf16, online softmax, async K/Q->LDS)
//   3) out = attn @ Wo + bo       (WMMA bf16 GEMM, f32 output)
// ---------------------------------------------------------------------------

typedef __bf16 bf16_t;
typedef __attribute__((ext_vector_type(16))) __bf16 v16bf;
typedef __attribute__((ext_vector_type(8)))  float   v8f;

__device__ __forceinline__ v8f wmma_bf16(v16bf a, v16bf b, v8f c) {
  return __builtin_amdgcn_wmma_f32_16x16x32_bf16(
      /*neg_a=*/false, a, /*neg_b=*/false, b,
      /*c_mod=*/(short)0, c, /*reuse_a=*/false, /*reuse_b=*/false);
}

// ---- gfx1250 async global->LDS DMA (ASYNCcnt-tracked), 16B per lane --------
// VDST VGPR holds the LDS byte offset; VADDR pair holds the 64-bit global
// address (GV mode, saddr = off).  Low 32 bits of a generic pointer to LDS
// are the LDS-relative offset.
__device__ __forceinline__ void async_ld_b128(void* lds, const void* gsrc) {
  unsigned int loff = (unsigned int)(uintptr_t)lds;
  asm volatile("global_load_async_to_lds_b128 %0, %1, off"
               :
               : "v"(loff), "v"(gsrc)
               : "memory");
}
__device__ __forceinline__ void wait_async0() {
  asm volatile("s_wait_asynccnt 0x0" ::: "memory");
}

// Build a 16-element bf16 fragment from two 16-byte LDS chunks.
__device__ __forceinline__ v16bf frag_pair(const bf16_t* p0, const bf16_t* p1) {
  union U { v16bf v; float4 q[2]; } u;
  u.q[0] = *(const float4*)p0;
  u.q[1] = *(const float4*)p1;
  return u.v;
}

// A-matrix fragment (16x32 bf16) from row-major LDS tile.
// ISA layout: lane L holds row (L&15); vec elems 0..7 -> K = k0 + 8*(L>=16) + 0..7,
// elems 8..15 -> same + 16.
__device__ __forceinline__ v16bf load_a_frag(const bf16_t* base, int row0,
                                             int stride, int k0) {
  const int lane = threadIdx.x & 31;
  const bf16_t* r = base + (row0 + (lane & 15)) * stride + k0 + ((lane >> 4) << 3);
  return frag_pair(r, r + 16);
}

// B-matrix fragment (32x16 bf16) from an N-major ("transposed") LDS tile:
// element (k, n) stored at base[n*stride + k].
// ISA layout: lane L holds col (L&15); vec elems 0..7 -> K = k0 + 8*(L>=16) + 0..7,
// elems 8..15 -> same + 16.
__device__ __forceinline__ v16bf load_b_frag(const bf16_t* base, int col0,
                                             int stride, int k0) {
  const int lane = threadIdx.x & 31;
  const bf16_t* r = base + (col0 + (lane & 15)) * stride + k0 + ((lane >> 4) << 3);
  return frag_pair(r, r + 16);
}

// ---------------------------------------------------------------------------
// Tiled GEMM: Y[M,N] = X[M,K] @ W[K,N] + bias.  X is f32 or bf16; W,bias f32.
// Block: 128 threads (4 waves) -> 64x64 output tile; wave -> 32x32 (2x2 WMMA).
// (Staging keeps the VGPR round-trip because of the f32->bf16 conversion.)
// ---------------------------------------------------------------------------
template <typename TIN, bool OUT_BF16>
__global__ __launch_bounds__(128)
void gemm_wmma(const TIN* __restrict__ X, const float* __restrict__ W,
               const float* __restrict__ bias, void* __restrict__ Yv,
               int M, int N, int K) {
  __shared__ __align__(16) bf16_t ldsA[64 * 32];   // row-major [row][k]
  __shared__ __align__(16) bf16_t ldsBt[64 * 32];  // n-major  [n][k]

  const int tid   = threadIdx.x;
  const int wave  = tid >> 5;
  const int lane  = tid & 31;
  const int half  = lane >> 4;
  const int mbase = blockIdx.x * 64;
  const int nbase = blockIdx.y * 64;
  const int rbase = (wave >> 1) * 32;  // wave's row offset inside 64
  const int cbase = (wave & 1) * 32;   // wave's col offset inside 64

  v8f acc[2][2] = {};

  for (int k0 = 0; k0 < K; k0 += 32) {
    __syncthreads();
    // Stage A: 64x32 elements, convert to bf16.
#pragma unroll
    for (int e = 0; e < 16; ++e) {
      int idx = e * 128 + tid;            // 0..2047
      int row = idx >> 5, kk = idx & 31;
      float v = (float)X[(size_t)(mbase + row) * K + k0 + kk];
      ldsA[row * 32 + kk] = (bf16_t)v;
    }
    // Stage W transposed: tile is 32(k) x 64(n) from W[K,N].
#pragma unroll
    for (int e = 0; e < 16; ++e) {
      int idx = e * 128 + tid;            // 0..2047
      int kk = idx >> 6, nn = idx & 63;
      float v = W[(size_t)(k0 + kk) * N + nbase + nn];
      ldsBt[nn * 32 + kk] = (bf16_t)v;
    }
    if (k0 + 32 < K)  // hint next A tile toward the caches
      __builtin_prefetch(&X[(size_t)mbase * K + k0 + 32], 0, 1);
    __syncthreads();

    v16bf a0 = load_a_frag(ldsA, rbase + 0, 32, 0);
    v16bf a1 = load_a_frag(ldsA, rbase + 16, 32, 0);
    v16bf b0 = load_b_frag(ldsBt, cbase + 0, 32, 0);
    v16bf b1 = load_b_frag(ldsBt, cbase + 16, 32, 0);
    acc[0][0] = wmma_bf16(a0, b0, acc[0][0]);
    acc[0][1] = wmma_bf16(a0, b1, acc[0][1]);
    acc[1][0] = wmma_bf16(a1, b0, acc[1][0]);
    acc[1][1] = wmma_bf16(a1, b1, acc[1][1]);
  }

  // Epilogue: C layout -> global.  Row = e + 8*half, col = lane&15.
#pragma unroll
  for (int i = 0; i < 2; ++i)
#pragma unroll
    for (int j = 0; j < 2; ++j)
#pragma unroll
      for (int e = 0; e < 8; ++e) {
        int row = mbase + rbase + i * 16 + e + half * 8;
        int col = nbase + cbase + j * 16 + (lane & 15);
        float v = acc[i][j][e] + bias[col];
        if (OUT_BF16)
          ((bf16_t*)Yv)[(size_t)row * N + col] = (bf16_t)v;
        else
          ((float*)Yv)[(size_t)row * N + col] = v;
      }
}

// ---------------------------------------------------------------------------
// Causal flash attention.  Q,K,V bf16 [B,S,128] -> O bf16 [B,S,128].
// Block: 128 threads (4 waves), 64 query rows per block, 16 rows per wave.
// Key loop in blocks of 32; online softmax; Q/K tiles DMA'd with
// GLOBAL_LOAD_ASYNC_TO_LDS_B128 (ASYNCcnt); V transposed manually.
// ---------------------------------------------------------------------------
__global__ __launch_bounds__(128)
void flash_attn(const bf16_t* __restrict__ Qg, const bf16_t* __restrict__ Kg,
                const bf16_t* __restrict__ Vg, bf16_t* __restrict__ Og,
                int S, float scale) {
  __shared__ __align__(16) bf16_t ldsQ[64 * 128];   // [qrow][d]
  __shared__ __align__(16) bf16_t ldsK[32 * 128];   // [key][d]   (B^T layout for S=QK^T)
  __shared__ __align__(16) bf16_t ldsVt[128 * 32];  // [d][key]   (B^T layout for O=PV)
  __shared__ __align__(16) bf16_t ldsP[4][16 * 32]; // per-wave P re-stripe patch

  const int tid   = threadIdx.x;
  const int wave  = tid >> 5;
  const int lane  = tid & 31;
  const int half  = lane >> 4;
  const int qbase = blockIdx.x * 64;
  const int batch = blockIdx.y;

  const bf16_t* Qb = Qg + (size_t)batch * S * 128;
  const bf16_t* Kb = Kg + (size_t)batch * S * 128;
  const bf16_t* Vb = Vg + (size_t)batch * S * 128;

  // DMA the 64x128 Q block (contiguous 16 KB) straight into LDS.
  {
    const char* src = (const char*)(Qb + (size_t)qbase * 128);
    char* dst = (char*)ldsQ;
#pragma unroll
    for (int e = 0; e < 8; ++e) {
      int off = (e * 128 + tid) * 16;
      async_ld_b128(dst + off, src + off);
    }
  }
  wait_async0();
  __syncthreads();

  // Preload this wave's Q A-fragments for the 4 K-chunks of d=128.
  v16bf qf[4];
#pragma unroll
  for (int dk = 0; dk < 4; ++dk)
    qf[dk] = load_a_frag(ldsQ, wave * 16, 128, dk * 32);

  v8f o[8] = {};               // 16 rows x 128 d accumulator (8 C tiles)
  float m[8], l[8];
#pragma unroll
  for (int e = 0; e < 8; ++e) { m[e] = -3.0e30f; l[e] = 0.0f; }

  const int nkb = qbase / 32 + 2;  // causal: keys 0 .. qbase+63
  for (int kb = 0; kb < nkb; ++kb) {
    const int kbeg = kb * 32;
    __syncthreads();  // previous iteration done reading ldsK/ldsVt
    // DMA the K block (32x128, contiguous 8 KB) straight into LDS.
    {
      const char* src = (const char*)(Kb + (size_t)kbeg * 128);
      char* dst = (char*)ldsK;
#pragma unroll
      for (int e = 0; e < 4; ++e) {
        int off = (e * 128 + tid) * 16;
        async_ld_b128(dst + off, src + off);
      }
    }
    // Stage V block transposed: [d][key] (needs data reshuffle -> manual).
    {
      const bf16_t* src = Vb + (size_t)kbeg * 128;
#pragma unroll
      for (int e = 0; e < 32; ++e) {
        int idx = e * 128 + tid;          // 0..4095
        int key = idx >> 7, d = idx & 127;
        ldsVt[d * 32 + key] = src[idx];
      }
    }
    wait_async0();
    __syncthreads();

    // S tiles: 16 rows x 32 keys, f32 accum over d=128 (4 chained WMMAs each).
    v8f c0 = {}, c1 = {};
#pragma unroll
    for (int dk = 0; dk < 4; ++dk) {
      v16bf b0 = load_b_frag(ldsK, 0, 128, dk * 32);   // keys kbeg+0..15
      v16bf b1 = load_b_frag(ldsK, 16, 128, dk * 32);  // keys kbeg+16..31
      c0 = wmma_bf16(qf[dk], b0, c0);
      c1 = wmma_bf16(qf[dk], b1, c1);
    }

    // Online softmax over this 32-key block.
    float fac[8];
#pragma unroll
    for (int e = 0; e < 8; ++e) {
      const int qrow = qbase + wave * 16 + e + half * 8;
      float s0 = c0[e] * scale;
      float s1 = c1[e] * scale;
      if (kbeg + (lane & 15) > qrow)      s0 = -3.0e30f;  // causal mask
      if (kbeg + 16 + (lane & 15) > qrow) s1 = -3.0e30f;
      float t = fmaxf(s0, s1);
#pragma unroll
      for (int off = 1; off < 16; off <<= 1) t = fmaxf(t, __shfl_xor(t, off, 32));
      const float mn = fmaxf(m[e], t);
      const float f  = __expf(m[e] - mn);
      const float p0 = __expf(s0 - mn);
      const float p1 = __expf(s1 - mn);
      float rs = p0 + p1;
#pragma unroll
      for (int off = 1; off < 16; off <<= 1) rs += __shfl_xor(rs, off, 32);
      l[e] = l[e] * f + rs;
      m[e] = mn;
      fac[e] = f;
      // C layout -> LDS patch (row-major 16x32) so it can be re-striped as A.
      const int prow = e + half * 8;
      ldsP[wave][prow * 32 + (lane & 15)]      = (bf16_t)p0;
      ldsP[wave][prow * 32 + 16 + (lane & 15)] = (bf16_t)p1;
    }
    // Rescale running O by exp(m_old - m_new), row-wise.
#pragma unroll
    for (int t = 0; t < 8; ++t)
#pragma unroll
      for (int e = 0; e < 8; ++e) o[t][e] *= fac[e];

    // Intra-wave LDS visibility for the P patch (cross-lane through memory).
    asm volatile("s_wait_dscnt 0" ::: "memory");

    // O(16x128) += P(16x32) @ V(32x128): one A-frag, 8 B-frags/WMMAs.
    v16bf pf = load_a_frag(ldsP[wave], 0, 32, 0);
#pragma unroll
    for (int t = 0; t < 8; ++t) {
      v16bf vf = load_b_frag(ldsVt, t * 16, 32, 0);
      o[t] = wmma_bf16(pf, vf, o[t]);
    }
  }

  // Normalize and store (C layout -> global bf16).
#pragma unroll
  for (int t = 0; t < 8; ++t)
#pragma unroll
    for (int e = 0; e < 8; ++e) {
      const int row = qbase + wave * 16 + e + half * 8;
      const int col = t * 16 + (lane & 15);
      const float inv = (l[e] > 0.0f) ? (1.0f / l[e]) : 0.0f;  // nan_to_num
      Og[((size_t)batch * S + row) * 128 + col] = (bf16_t)(o[t][e] * inv);
    }
}

// ---------------------------------------------------------------------------
// Launch wrapper.
// ---------------------------------------------------------------------------
extern "C" void kernel_launch(void* const* d_in, const int* in_sizes, int n_in,
                              void* d_out, int out_size, void* d_ws, size_t ws_size,
                              hipStream_t stream) {
  (void)in_sizes; (void)n_in; (void)out_size; (void)ws_size;

  const float* query = (const float*)d_in[0];
  const float* key   = (const float*)d_in[1];
  const float* value = (const float*)d_in[2];
  // d_in[3] = causal mask (int32) -- reproduced analytically, not read.
  const float* Wq = (const float*)d_in[4];
  const float* bq = (const float*)d_in[5];
  const float* Wk = (const float*)d_in[6];
  const float* bk = (const float*)d_in[7];
  const float* Wv = (const float*)d_in[8];
  const float* bv = (const float*)d_in[9];
  const float* Wo = (const float*)d_in[10];
  const float* bo = (const float*)d_in[11];
  float* out = (float*)d_out;

  const int Bc = 8, Sc = 2048, Ec = 1024, Dc = 128, Oc = 1024;
  const int M = Bc * Sc;  // 16384

  // Workspace: Q,K,V,attn-out in bf16 -> 4 * 16384 * 128 * 2 = 16 MB.
  bf16_t* Qbf = (bf16_t*)d_ws;
  bf16_t* Kbf = Qbf + (size_t)M * Dc;
  bf16_t* Vbf = Kbf + (size_t)M * Dc;
  bf16_t* Abf = Vbf + (size_t)M * Dc;

  dim3 blk(128);

  // 1) QKV projections.
  dim3 gproj(M / 64, Dc / 64);
  gemm_wmma<float, true><<<gproj, blk, 0, stream>>>(query, Wq, bq, Qbf, M, Dc, Ec);
  gemm_wmma<float, true><<<gproj, blk, 0, stream>>>(key,   Wk, bk, Kbf, M, Dc, Ec);
  gemm_wmma<float, true><<<gproj, blk, 0, stream>>>(value, Wv, bv, Vbf, M, Dc, Ec);

  // 2) Causal flash attention.
  dim3 gattn(Sc / 64, Bc);
  flash_attn<<<gattn, blk, 0, stream>>>(Qbf, Kbf, Vbf, Abf, Sc,
                                        1.0f / sqrtf((float)Dc));

  // 3) Output projection (f32 result).
  dim3 gout(M / 64, Oc / 64);
  gemm_wmma<bf16_t, false><<<gout, blk, 0, stream>>>(Abf, Wo, bo, out, M, Oc, Dc);
}
